// NonLocalBlock_13168369730190
// MI455X (gfx1250) — matrix operand order repaired and checked by reading, not compiled, not run on hardware
//
#include <hip/hip_runtime.h>

// ---------------------------------------------------------------------------
// NonLocal block for MI455X (gfx1250): all GEMMs via v_wmma_f32_16x16x32_bf16,
// A-tile staging via global_load_async_to_lds_b128 (ASYNCcnt path).
// Pipeline:
//   k_init        : weights fp32->bf16 (w_mask transposed), colsum=0
//   k_convert_x   : x fp32 -> bf16
//   k_gemm_qkv    : phi[o][m], thetaT[n][o], gT[m][o]  (theta/g stored transposed)
//   k_gemm_attn   : e[n][m]=exp(thetaT·phi) in bf16 + atomic column sums over n
//   k_scale_g     : gsT[m][c] = gT[m][c] / colsum[m]   (fold softmax denominator)
//   k_gemm_ag     : AGT[n][c] = e · gsT
//   k_gemm_mask   : out[co][n] = (AGT · w_maskT)[n][co] + x[co][n]  (residual fused)
// ---------------------------------------------------------------------------

typedef __attribute__((ext_vector_type(16))) __bf16 v16bf;
typedef __attribute__((ext_vector_type(8)))  float  v8f;

union FragAB { v16bf v; unsigned u[8]; };

#define BM 128
#define BN 128
#define BK 64
#define LSTR 72          // LDS row stride in ushorts: 144B (16B multiple)
#define NTHREADS 256

static constexpr int Bb  = 4;
static constexpr int Cc  = 256;
static constexpr int HID = 128;
static constexpr int Nsp = 4096;   // H*W

__device__ __forceinline__ unsigned short f2bf(float f) {
  unsigned u = __float_as_uint(f);
  return (unsigned short)((u + 0x7FFFu + ((u >> 16) & 1u)) >> 16);   // RNE
}
__device__ __forceinline__ float bf2f(unsigned short h) {
  return __uint_as_float(((unsigned)h) << 16);
}

// ---------------------------------------------------------------------------
// Shared tiled WMMA GEMM core: C[128x128] block, A row-major [M][K],
// B row-major [K][N], bf16 operands, f32 accumulation. 8 waves; each wave
// computes a 32x64 sub-tile as 2x4 fragments of 16x16.
// A tile is staged memory->LDS with async copies (no VGPR round-trip);
// B tile is transposed through registers (transpose has to touch VGPRs).
// ---------------------------------------------------------------------------
__device__ __forceinline__ void gemm_block(
    unsigned short* As, unsigned short* Bs,
    const unsigned short* __restrict__ A, int lda,
    const unsigned short* __restrict__ B, int ldb,
    int K, int rowBase, int colBase, v8f acc[2][4])
{
  const int tid  = threadIdx.x;
  const int lane = tid & 31;
  const int wave = tid >> 5;
  const int wm   = wave >> 1;          // 0..3 (row block of wave)
  const int wn   = wave & 1;           // 0..1 (col block of wave)
  const int half = (lane >> 4) & 1;
  const int l15  = lane & 15;

  for (int k0 = 0; k0 < K; k0 += BK) {
    __syncthreads();                   // prior iteration's readers are done
    if (k0 + BK < K) {                 // warm next B tile in near caches
      __builtin_prefetch(B + (size_t)(k0 + BK + (tid & 63)) * ldb + colBase, 0, 3);
    }
    // A tile: 128 x 64 bf16, async global->LDS copy (tracked on ASYNCcnt)
    #pragma unroll
    for (int v = tid; v < (BM * BK) / 8; v += NTHREADS) {
      int r  = v >> 3;
      int kc = (v & 7) * 8;
      unsigned           lva = (unsigned)(size_t)(As + r * LSTR + kc);
      unsigned long long gva = (unsigned long long)(size_t)
                               (A + (size_t)(rowBase + r) * lda + k0 + kc);
      asm volatile("global_load_async_to_lds_b128 %0, %1, off"
                   :: "v"(lva), "v"(gva) : "memory");
    }
    // B tile: 64 x 128 bf16, transposed into Bs[n][k] through registers
    #pragma unroll
    for (int v = tid; v < (BK * BN) / 8; v += NTHREADS) {
      int kr = v >> 4;
      int cc = (v & 15) * 8;
      uint4 d = *(const uint4*)(B + (size_t)(k0 + kr) * ldb + colBase + cc);
      const unsigned short* s = (const unsigned short*)&d;
      #pragma unroll
      for (int j = 0; j < 8; ++j) Bs[(cc + j) * LSTR + kr] = s[j];
    }
    asm volatile("s_wait_asynccnt 0x0" ::: "memory");   // A tile landed in LDS
    __syncthreads();

    #pragma unroll
    for (int kk = 0; kk < BK; kk += 32) {
      FragAB a[2], bq[4];
      // A fragment, 16x32 bf16 layout (ISA 7.12.2): lane half selects K block
      #pragma unroll
      for (int mf = 0; mf < 2; ++mf) {
        int row = wm * 32 + mf * 16 + l15;
        #pragma unroll
        for (int i = 0; i < 8; ++i) {
          int kpos = kk + 16 * (i >> 2) + 8 * half + 2 * (i & 3);
          a[mf].u[i] = *(const unsigned*)(As + row * LSTR + kpos);
        }
      }
      // B fragment, 32x16 bf16: same K order as A (lane half selects 8-wide K block)
      #pragma unroll
      for (int nf = 0; nf < 4; ++nf) {
        int col = wn * 64 + nf * 16 + l15;
        #pragma unroll
        for (int i = 0; i < 8; ++i) {
          int kpos = kk + 16 * (i >> 2) + 8 * half + 2 * (i & 3);
          bq[nf].u[i] = *(const unsigned*)(Bs + col * LSTR + kpos);
        }
      }
      #pragma unroll
      for (int mf = 0; mf < 2; ++mf)
        #pragma unroll
        for (int nf = 0; nf < 4; ++nf)
          acc[mf][nf] = __builtin_amdgcn_wmma_f32_16x16x32_bf16(
              false, a[mf].v, false, bq[nf].v, (short)0, acc[mf][nf], false, false);
    }
  }
}

#define GEMM_PROLOGUE()                                   \
  __shared__ unsigned short As[BM * LSTR];                \
  __shared__ unsigned short Bs[BN * LSTR];                \
  v8f acc[2][4];                                          \
  _Pragma("unroll") for (int mf = 0; mf < 2; ++mf)        \
    _Pragma("unroll") for (int nf = 0; nf < 4; ++nf)      \
      _Pragma("unroll") for (int r = 0; r < 8; ++r)       \
        acc[mf][nf][r] = 0.0f;                            \
  const int tid  = threadIdx.x;                           \
  const int lane = tid & 31;                              \
  const int wave = tid >> 5;                              \
  const int wm = wave >> 1, wn = wave & 1;                \
  const int half = (lane >> 4) & 1, l15 = lane & 15;      \
  (void)lane;

// ---------------------------------------------------------------------------
// Small kernels
// ---------------------------------------------------------------------------
__global__ void k_init(const float* __restrict__ wphi, const float* __restrict__ wtheta,
                       const float* __restrict__ wg,  const float* __restrict__ wmask,
                       unsigned short* __restrict__ wbf, unsigned short* __restrict__ wmaskT,
                       float* __restrict__ colsum)
{
  int idx = blockIdx.x * NTHREADS + threadIdx.x;
  if (idx < 3 * HID * Cc) {
    const float* src = (idx < HID * Cc) ? wphi : (idx < 2 * HID * Cc ? wtheta : wg);
    wbf[idx] = f2bf(src[idx % (HID * Cc)]);
  }
  if (idx < HID * Cc) {                 // w_mask [C][HID] -> wmaskT [HID][C]
    int h = idx >> 8, co = idx & 255;
    wmaskT[idx] = f2bf(wmask[co * HID + h]);
  }
  if (idx < Bb * Nsp) colsum[idx] = 0.0f;
}

__global__ void k_convert_x(const float* __restrict__ x, unsigned short* __restrict__ xbf)
{
  int i = blockIdx.x * NTHREADS + threadIdx.x;     // one float4 per thread
  float4 v = ((const float4*)x)[i];
  ushort4 o;
  o.x = f2bf(v.x); o.y = f2bf(v.y); o.z = f2bf(v.z); o.w = f2bf(v.w);
  *(ushort4*)(xbf + 4 * i) = o;
}

__global__ void k_scale_g(const unsigned short* __restrict__ gT,
                          const float* __restrict__ colsum,
                          unsigned short* __restrict__ gsT)
{
  int idx = blockIdx.x * NTHREADS + threadIdx.x;   // B*Nsp*HID elements
  int b = idx / (Nsp * HID);
  int m = (idx / HID) % Nsp;
  gsT[idx] = f2bf(bf2f(gT[idx]) / colsum[b * Nsp + m]);
}

// ---------------------------------------------------------------------------
// GEMM kernels
// ---------------------------------------------------------------------------
__global__ void __launch_bounds__(NTHREADS)
k_gemm_qkv(const unsigned short* __restrict__ wbf, const unsigned short* __restrict__ xbf,
           unsigned short* __restrict__ phi, unsigned short* __restrict__ thetaT,
           unsigned short* __restrict__ gT)
{
  const int b = blockIdx.z / 3, sel = blockIdx.z % 3;
  const unsigned short* A = wbf + sel * (HID * Cc);                // [128][256]
  const unsigned short* Bm = xbf + (size_t)b * Cc * Nsp;           // [256][4096]
  const int rowBase = 0, colBase = blockIdx.x * BN;
  GEMM_PROLOGUE();
  gemm_block(As, Bs, A, Cc, Bm, Nsp, Cc, rowBase, colBase, acc);

  if (sel == 0) {
    unsigned short* out = phi + (size_t)b * HID * Nsp;             // [o][m]
    #pragma unroll
    for (int mf = 0; mf < 2; ++mf)
      #pragma unroll
      for (int nf = 0; nf < 4; ++nf)
        #pragma unroll
        for (int r = 0; r < 8; ++r) {
          int row = rowBase + wm * 32 + mf * 16 + half * 8 + r;
          int col = colBase + wn * 64 + nf * 16 + l15;
          out[(size_t)row * Nsp + col] = f2bf(acc[mf][nf][r]);
        }
  } else {
    unsigned short* out = (sel == 1 ? thetaT : gT) + (size_t)b * Nsp * HID;  // [n|m][o]
    #pragma unroll
    for (int mf = 0; mf < 2; ++mf)
      #pragma unroll
      for (int nf = 0; nf < 4; ++nf)
        #pragma unroll
        for (int r = 0; r < 8; ++r) {
          int row = rowBase + wm * 32 + mf * 16 + half * 8 + r;
          int col = colBase + wn * 64 + nf * 16 + l15;
          out[(size_t)col * HID + row] = f2bf(acc[mf][nf][r]);
        }
  }
}

__global__ void __launch_bounds__(NTHREADS)
k_gemm_attn(const unsigned short* __restrict__ thetaT, const unsigned short* __restrict__ phi,
            unsigned short* __restrict__ Abf, float* __restrict__ colsum)
{
  const int b = blockIdx.z;
  const unsigned short* A  = thetaT + (size_t)b * Nsp * HID;       // [4096][128]
  const unsigned short* Bm = phi    + (size_t)b * HID * Nsp;       // [128][4096]
  const int rowBase = blockIdx.y * BM, colBase = blockIdx.x * BN;
  GEMM_PROLOGUE();
  gemm_block(As, Bs, A, HID, Bm, Nsp, HID, rowBase, colBase, acc);

  // Epilogue: probabilities e = exp(score) in bf16 + atomic column sums (over n).
  unsigned short* out = Abf + (size_t)b * Nsp * Nsp;
  float csum[4] = {0.f, 0.f, 0.f, 0.f};
  #pragma unroll
  for (int mf = 0; mf < 2; ++mf)
    #pragma unroll
    for (int nf = 0; nf < 4; ++nf)
      #pragma unroll
      for (int r = 0; r < 8; ++r) {
        int row = rowBase + wm * 32 + mf * 16 + half * 8 + r;
        int col = colBase + wn * 64 + nf * 16 + l15;
        float e = __expf(acc[mf][nf][r]);       // logits bounded -> no max needed
        out[(size_t)row * Nsp + col] = f2bf(e);
        csum[nf] += e;                          // this lane's column partial
      }
  #pragma unroll
  for (int nf = 0; nf < 4; ++nf) {
    float s = csum[nf] + __shfl_xor(csum[nf], 16, 32);   // combine half-wave rows
    if (lane < 16)
      atomicAdd(&colsum[b * Nsp + colBase + wn * 64 + nf * 16 + lane], s);
  }
}

__global__ void __launch_bounds__(NTHREADS)
k_gemm_ag(const unsigned short* __restrict__ Abf, const unsigned short* __restrict__ gsT,
          unsigned short* __restrict__ AGT)
{
  const int b = blockIdx.z;
  const unsigned short* A  = Abf + (size_t)b * Nsp * Nsp;          // [4096][4096]
  const unsigned short* Bm = gsT + (size_t)b * Nsp * HID;          // [4096][128]
  const int rowBase = blockIdx.y * BM, colBase = 0;                // N = 128
  GEMM_PROLOGUE();
  gemm_block(As, Bs, A, Nsp, Bm, HID, Nsp, rowBase, colBase, acc);

  unsigned short* out = AGT + (size_t)b * Nsp * HID;               // [n][c]
  #pragma unroll
  for (int mf = 0; mf < 2; ++mf)
    #pragma unroll
    for (int nf = 0; nf < 4; ++nf)
      #pragma unroll
      for (int r = 0; r < 8; ++r) {
        int row = rowBase + wm * 32 + mf * 16 + half * 8 + r;
        int col = colBase + wn * 64 + nf * 16 + l15;
        out[(size_t)row * HID + col] = f2bf(acc[mf][nf][r]);
      }
}

__global__ void __launch_bounds__(NTHREADS)
k_gemm_mask(const unsigned short* __restrict__ AGT, const unsigned short* __restrict__ wmaskT,
            const float* __restrict__ x, float* __restrict__ outp)
{
  const int b = blockIdx.z;
  const unsigned short* A  = AGT + (size_t)b * Nsp * HID;          // [4096][128]
  const unsigned short* Bm = wmaskT;                               // [128][256]
  const int rowBase = blockIdx.y * BM, colBase = blockIdx.x * BN;  // cols = channels
  GEMM_PROLOGUE();
  gemm_block(As, Bs, A, HID, Bm, Cc, HID, rowBase, colBase, acc);

  // outT store + residual: out[b][co][n] = C[n][co] + x[b][co][n]
  const float* xb = x    + (size_t)b * Cc * Nsp;
  float*       ob = outp + (size_t)b * Cc * Nsp;
  #pragma unroll
  for (int mf = 0; mf < 2; ++mf)
    #pragma unroll
    for (int nf = 0; nf < 4; ++nf)
      #pragma unroll
      for (int r = 0; r < 8; ++r) {
        int n  = rowBase + wm * 32 + mf * 16 + half * 8 + r;
        int co = colBase + wn * 64 + nf * 16 + l15;
        size_t idx = (size_t)co * Nsp + n;
        ob[idx] = acc[mf][nf][r] + xb[idx];
      }
}

// ---------------------------------------------------------------------------
// Host launcher
// ---------------------------------------------------------------------------
extern "C" void kernel_launch(void* const* d_in, const int* in_sizes, int n_in,
                              void* d_out, int out_size, void* d_ws, size_t ws_size,
                              hipStream_t stream) {
  const float* x      = (const float*)d_in[0];
  const float* wphi   = (const float*)d_in[1];
  const float* wtheta = (const float*)d_in[2];
  const float* wg     = (const float*)d_in[3];
  const float* wmask  = (const float*)d_in[4];
  float* outp = (float*)d_out;

  // Workspace carve-out (all sizes already 256B-aligned)
  char* p = (char*)d_ws;
  auto take = [&](size_t bytes) { char* r = p; p += bytes; return r; };
  unsigned short* xbf    = (unsigned short*)take((size_t)Bb * Cc * Nsp * 2);    //  8 MB
  unsigned short* wbf    = (unsigned short*)take((size_t)3 * HID * Cc * 2);
  unsigned short* wmaskT = (unsigned short*)take((size_t)HID * Cc * 2);
  unsigned short* phi    = (unsigned short*)take((size_t)Bb * HID * Nsp * 2);   //  4 MB
  unsigned short* thetaT = (unsigned short*)take((size_t)Bb * Nsp * HID * 2);
  unsigned short* gT     = (unsigned short*)take((size_t)Bb * Nsp * HID * 2);
  unsigned short* gsT    = (unsigned short*)take((size_t)Bb * Nsp * HID * 2);
  unsigned short* AGT    = (unsigned short*)take((size_t)Bb * Nsp * HID * 2);
  float*          colsum = (float*)take((size_t)Bb * Nsp * 4);
  unsigned short* Abf    = (unsigned short*)take((size_t)Bb * Nsp * Nsp * 2);   // 128 MB

  k_init<<<(3 * HID * Cc + NTHREADS - 1) / NTHREADS, NTHREADS, 0, stream>>>(
      wphi, wtheta, wg, wmask, wbf, wmaskT, colsum);

  k_convert_x<<<(Bb * Cc * Nsp / 4) / NTHREADS, NTHREADS, 0, stream>>>(x, xbf);

  dim3 gq(Nsp / BN, 1, Bb * 3);
  k_gemm_qkv<<<gq, NTHREADS, 0, stream>>>(wbf, xbf, phi, thetaT, gT);

  dim3 ga(Nsp / BN, Nsp / BM, Bb);
  k_gemm_attn<<<ga, NTHREADS, 0, stream>>>(thetaT, phi, Abf, colsum);

  k_scale_g<<<(Bb * Nsp * HID) / NTHREADS, NTHREADS, 0, stream>>>(gT, colsum, gsT);

  dim3 gg(1, Nsp / BM, Bb);
  k_gemm_ag<<<gg, NTHREADS, 0, stream>>>(Abf, gsT, AGT);

  dim3 gm(Cc / BN, Nsp / BM, Bb);
  k_gemm_mask<<<gm, NTHREADS, 0, stream>>>(AGT, wmaskT, x, outp);
}
